// LinearAttention_53730040873581
// MI455X (gfx1250) — matrix operand order repaired and checked by reading, not compiled, not run on hardware
//
#include <hip/hip_runtime.h>

#define DEV __device__ __forceinline__

typedef __bf16 bf16;
typedef __attribute__((ext_vector_type(16))) bf16 v16bf;
typedef __attribute__((ext_vector_type(8)))  bf16 bf16x8;
typedef __attribute__((ext_vector_type(4)))  bf16 bf16x4;
typedef __attribute__((ext_vector_type(8)))  float v8f;
typedef __attribute__((ext_vector_type(4)))  float f32x4;

static constexpr int   SEQ = 2048;
static constexpr int   DM  = 1024;
static constexpr int   NH  = 16;
static constexpr int   DH  = 64;
static constexpr float EPS = 1e-6f;

DEV v8f wmma_bf16(v16bf a, v16bf b, v8f c) {
  // D = A(16x32 bf16) x B(32x16 bf16) + C(16x16 f32)
  return __builtin_amdgcn_wmma_f32_16x16x32_bf16(false, a, false, b, (short)0, c,
                                                 false, false);
}

DEV v16bf load_2x8(const bf16* p, int off2) {
  union { v16bf v; bf16x8 h[2]; } u;
  u.h[0] = *(const bf16x8*)(p);
  u.h[1] = *(const bf16x8*)(p + off2);
  return u.v;
}

// A-operand fragment (16xK=32) from row-major tile[m][k], rows mbase..mbase+15.
// Lane<16 holds K = {0..7, 16..23}+32*ks ; lane>=16 holds K = {8..15, 24..31}+32*ks.
DEV v16bf frag_a(const bf16* t, int ldm, int mbase, int ks, int L) {
  const bf16* p = t + (mbase + (L & 15)) * ldm + ks * 32 + ((L < 16) ? 0 : 8);
  return load_2x8(p, 16);
}
// B-operand fragment (K=32 x 16) from tile stored as tile[col][k] (K contiguous).
// Lane<16: col=L,    K = {0..7, 16..23} (+32*ks); lane>=16: col=L-16, K = {8..15, 24..31} (+32*ks).
DEV v16bf frag_b(const bf16* t, int ldm, int cbase, int ks, int L) {
  const bf16* p = t + (cbase + (L & 15)) * ldm + ks * 32 + ((L < 16) ? 0 : 8);
  return load_2x8(p, 16);
}

DEV void split_bf16x4(f32x4 v, bf16x4& hi, bf16x4& lo) {
#pragma unroll
  for (int j = 0; j < 4; ++j) {
    bf16 h = (bf16)v[j];
    hi[j] = h;
    lo[j] = (bf16)(v[j] - (float)h);
  }
}

// C = act( A @ W^T + bias ), bf16x3 error-compensated fp32 GEMM.
// Block tile 128x64xBK64, 8 waves (4x2), each wave owns 2x2 16x16 accumulators
// (24 WMMAs per wave per K-iteration between one barrier pair).
//   A: [M,K] row-major, or head-major [K/64][M][64] when AHM (ldA becomes 64)
//   W: [Nout,K] row-major (column j of W^T is row j of W, K contiguous)
// out: row-major [M,Nout], or head-major [Nout/64][M][64] when OHM
template <int AHM, int ACT, int OHM>
__global__ __launch_bounds__(256) void gemm_bias_act(
    const float* __restrict__ A, const float* __restrict__ W,
    const float* __restrict__ bias, float* __restrict__ out,
    int M, int K, int Nout) {
  __shared__ __align__(16) bf16 Ah[128][72];
  __shared__ __align__(16) bf16 Al[128][72];
  __shared__ __align__(16) bf16 Bh[64][72];
  __shared__ __align__(16) bf16 Bl[64][72];

  const int m0 = blockIdx.x * 128, n0 = blockIdx.y * 64;
  const int tid = threadIdx.x;
  const int w = tid >> 5, L = tid & 31;
  const int wm = w & 3, wn = w >> 2;   // wave covers rows 32*wm.., cols 32*wn..
  const int q4 = (tid & 15) * 4;       // float4 slot within a 64-wide K slice
  const int r0 = tid >> 4;             // 0..15
  const long ldA = AHM ? 64 : (long)K;

  v8f acc[2][2] = {};

  for (int k0 = 0; k0 < K; k0 += 64) {
    // With BK == 64 == head stride, head-major A is a plain strided tile.
    const float* abase = AHM ? A + (long)(k0 >> 6) * M * 64 + (long)m0 * 64 + q4
                             : A + (long)m0 * K + k0 + q4;
    const float* wbase = W + (long)n0 * K + k0 + q4;

    // Stage A: 128x64 f32 -> bf16 hi/lo (b128 loads, b64 stores).
#pragma unroll
    for (int rr = 0; rr < 8; ++rr) {
      int r = r0 + 16 * rr;
      f32x4 av = *(const f32x4*)(abase + (long)r * ldA);
      bf16x4 hv, lv;
      split_bf16x4(av, hv, lv);
      *(bf16x4*)&Ah[r][q4] = hv;
      *(bf16x4*)&Al[r][q4] = lv;
    }
    // Stage W: 64x64.
#pragma unroll
    for (int rr = 0; rr < 4; ++rr) {
      int r = r0 + 16 * rr;
      f32x4 wv = *(const f32x4*)(wbase + (long)r * K);
      bf16x4 hv, lv;
      split_bf16x4(wv, hv, lv);
      *(bf16x4*)&Bh[r][q4] = hv;
      *(bf16x4*)&Bl[r][q4] = lv;
    }
    // Cooperative prefetch of the next K slice (one cacheline per thread).
    if (k0 + 64 < K) {
      const float* anext = AHM ? A + (long)((k0 >> 6) + 1) * M * 64 + (long)m0 * 64
                               : A + (long)m0 * K + k0 + 64;
      __builtin_prefetch(anext + (long)(tid >> 1) * ldA + (tid & 1) * 32, 0, 3);
      if (tid < 128)
        __builtin_prefetch(W + (long)(n0 + (tid >> 1)) * K + k0 + 64 + (tid & 1) * 32,
                           0, 3);
    }
    __syncthreads();

#pragma unroll
    for (int ks = 0; ks < 2; ++ks) {
      v16bf ah[2], al[2], bh[2], bl[2];
#pragma unroll
      for (int mt = 0; mt < 2; ++mt) {
        ah[mt] = frag_a(&Ah[0][0], 72, 32 * wm + 16 * mt, ks, L);
        al[mt] = frag_a(&Al[0][0], 72, 32 * wm + 16 * mt, ks, L);
      }
#pragma unroll
      for (int nt = 0; nt < 2; ++nt) {
        bh[nt] = frag_b(&Bh[0][0], 72, 32 * wn + 16 * nt, ks, L);
        bl[nt] = frag_b(&Bl[0][0], 72, 32 * wn + 16 * nt, ks, L);
      }
      // bf16x3: HiHi + HiLo + LoHi  (~fp32-accurate, f32 accumulate)
#pragma unroll
      for (int mt = 0; mt < 2; ++mt)
#pragma unroll
        for (int nt = 0; nt < 2; ++nt) {
          acc[mt][nt] = wmma_bf16(ah[mt], bh[nt], acc[mt][nt]);
          acc[mt][nt] = wmma_bf16(ah[mt], bl[nt], acc[mt][nt]);
          acc[mt][nt] = wmma_bf16(al[mt], bh[nt], acc[mt][nt]);
        }
    }
    __syncthreads();
  }

  // Epilogue: lanes of a half-wave cover 16 consecutive columns -> 64B bursts.
#pragma unroll
  for (int mt = 0; mt < 2; ++mt)
#pragma unroll
    for (int nt = 0; nt < 2; ++nt) {
      int col = n0 + 32 * wn + 16 * nt + (L & 15);
      float bv = bias[col];
#pragma unroll
      for (int r = 0; r < 8; ++r) {
        int m = m0 + 32 * wm + 16 * mt + ((L < 16) ? r : r + 8);
        float v = acc[mt][nt][r] + bv;
        if (ACT) v = (v > 0.f) ? (v + 1.f) : __expf(v);  // elu(x)+1
        if (OHM)
          out[(long)(col >> 6) * M * 64 + (long)m * 64 + (col & 63)] = v;
        else
          out[(long)m * Nout + col] = v;
      }
    }
}

// Chunked causal linear attention, one block (8 waves) per head.
// Per chunk of 64: Am = tril(Q K^T) (computed transposed as K·Q^T so the LDS
// spill is row-contiguous); numer = Am@V + Q@S_prev; denom = rowsum(Am)+Q.z;
// then S += K^T@V (state kept in f32 WMMA accumulators across the sequence).
__global__ __launch_bounds__(256) void linattn_causal(
    const float* __restrict__ Qh, const float* __restrict__ Kh,
    const float* __restrict__ Vh, float* __restrict__ Oh) {
  __shared__ __align__(16) bf16 Qb[64][72];  // Q[n][d]
  __shared__ __align__(16) bf16 Kn[64][72];  // K[n][d]
  __shared__ __align__(16) bf16 Kt[64][72];  // K^T[d][n] (A for state update)
  __shared__ __align__(16) bf16 Vt[64][72];  // V^T[e][n] (B operand)
  __shared__ __align__(16) bf16 Am[64][72];  // tril(QK^T)[i][j]
  __shared__ __align__(16) bf16 St[64][72];  // S^T[e][d] (B for Q@S)
  __shared__ float zv[64];
  __shared__ float dn[64];

  const int h = blockIdx.x;
  const long base = (long)h * SEQ * DH;
  const int tid = threadIdx.x;
  const int w = tid >> 5, L = tid & 31;
  const int wm = w & 3, wn = w >> 2;
  const int tc0 = 32 * wn, tc1 = tc0 + 16;
  const int rb = (L < 16) ? 0 : 8;     // row offset of this lane's C elements

  v8f s0 = {}, s1 = {};  // this wave's two 16x16 tiles of S [d][e], f32
  if (tid < 64) zv[tid] = 0.f;
  __syncthreads();

  for (int cblk = 0; cblk < SEQ / 64; ++cblk) {
    const long cb = base + (long)cblk * 64 * DH;
    // Stage chunk (vectorized f32 -> bf16) and previous state S -> St^T.
#pragma unroll
    for (int i = 0; i < 4; ++i) {
      int idx4 = tid + 256 * i;     // 0..1023 float4s (4096 floats per tensor)
      int r = idx4 >> 4;            // 0..63
      int c4 = (idx4 & 15) * 4;
      long off = cb + (long)idx4 * 4;
      f32x4 q4 = *(const f32x4*)(Qh + off);
      f32x4 k4 = *(const f32x4*)(Kh + off);
      f32x4 v4 = *(const f32x4*)(Vh + off);
      bf16x4 qv, kv;
#pragma unroll
      for (int j = 0; j < 4; ++j) { qv[j] = (bf16)q4[j]; kv[j] = (bf16)k4[j]; }
      *(bf16x4*)&Qb[r][c4] = qv;
      *(bf16x4*)&Kn[r][c4] = kv;
#pragma unroll
      for (int j = 0; j < 4; ++j) {
        Kt[c4 + j][r] = kv[j];
        Vt[c4 + j][r] = (bf16)v4[j];
      }
      if (cblk + 1 < SEQ / 64) {
        __builtin_prefetch(Qh + off + 64 * DH, 0, 3);
        __builtin_prefetch(Kh + off + 64 * DH, 0, 3);
        __builtin_prefetch(Vh + off + 64 * DH, 0, 3);
      }
    }
    // S accum tile [d][e]: lane owns fixed e, contiguous d -> one b128 store.
#pragma unroll
    for (int nt = 0; nt < 2; ++nt) {
      v8f s = nt ? s1 : s0;
      int e = (nt ? tc1 : tc0) + (L & 15);
      union { bf16x8 v; bf16 x[8]; } pk;
#pragma unroll
      for (int r = 0; r < 8; ++r) pk.x[r] = (bf16)s[r];
      *(bf16x8*)&St[e][16 * wm + rb] = pk.v;
    }
    __syncthreads();

    // Am^T tiles = K·Q^T: accumulator [j][i]; lane owns fixed i (query col) and
    // 8 contiguous j (key rows) -> masked spill is one b128 store per tile.
    v8f a0 = {}, a1 = {};
#pragma unroll
    for (int ks = 0; ks < 2; ++ks) {
      v16bf ka = frag_a(&Kn[0][0], 72, 16 * wm, ks, L);
      a0 = wmma_bf16(ka, frag_b(&Qb[0][0], 72, tc0, ks, L), a0);
      a1 = wmma_bf16(ka, frag_b(&Qb[0][0], 72, tc1, ks, L), a1);
    }
#pragma unroll
    for (int nt = 0; nt < 2; ++nt) {
      v8f a = nt ? a1 : a0;
      int i = (nt ? tc1 : tc0) + (L & 15);
      int jb = 16 * wm + rb;
      union { bf16x8 v; bf16 x[8]; } pk;
#pragma unroll
      for (int r = 0; r < 8; ++r) {
        int j = jb + r;
        pk.x[r] = (bf16)((j <= i) ? a[r] : 0.f);   // causal tril, inclusive
      }
      *(bf16x8*)&Am[i][jb] = pk.v;
    }
    __syncthreads();

    // denom_i = eps + rowsum(Am_i) + q_i . z_prev
    if (tid < 64) {
      float s = EPS;
      for (int j = 0; j < 64; ++j) s += (float)Am[tid][j];
      for (int d = 0; d < 64; ++d) s += (float)Qb[tid][d] * zv[d];
      dn[tid] = s;
    }
    __syncthreads();

    // numer = Am @ V + Q @ S_prev ; out = numer / denom
    // (original orientation: global stores coalesce 64B across each half-wave)
    v8f o0 = {}, o1 = {};
#pragma unroll
    for (int ks = 0; ks < 2; ++ks) {
      v16bf aa = frag_a(&Am[0][0], 72, 16 * wm, ks, L);
      o0 = wmma_bf16(aa, frag_b(&Vt[0][0], 72, tc0, ks, L), o0);
      o1 = wmma_bf16(aa, frag_b(&Vt[0][0], 72, tc1, ks, L), o1);
    }
#pragma unroll
    for (int ks = 0; ks < 2; ++ks) {
      v16bf qa = frag_a(&Qb[0][0], 72, 16 * wm, ks, L);
      o0 = wmma_bf16(qa, frag_b(&St[0][0], 72, tc0, ks, L), o0);
      o1 = wmma_bf16(qa, frag_b(&St[0][0], 72, tc1, ks, L), o1);
    }
#pragma unroll
    for (int nt = 0; nt < 2; ++nt) {
      v8f o = nt ? o1 : o0;
      int e = (nt ? tc1 : tc0) + (L & 15);
#pragma unroll
      for (int r = 0; r < 8; ++r) {
        int i = 16 * wm + rb + r;
        Oh[cb + (long)i * DH + e] = o[r] / dn[i];
      }
    }

    // State update: S += K^T @ V (f32 accumulators), z += colsum(K).
#pragma unroll
    for (int ks = 0; ks < 2; ++ks) {
      v16bf ka = frag_a(&Kt[0][0], 72, 16 * wm, ks, L);
      s0 = wmma_bf16(ka, frag_b(&Vt[0][0], 72, tc0, ks, L), s0);
      s1 = wmma_bf16(ka, frag_b(&Vt[0][0], 72, tc1, ks, L), s1);
    }
    if (tid < 64) {
      float s = 0.f;
      for (int n = 0; n < 64; ++n) s += (float)Kt[tid][n];
      zv[tid] += s;
    }
    __syncthreads();
  }
}

extern "C" void kernel_launch(void* const* d_in, const int* in_sizes, int n_in,
                              void* d_out, int out_size, void* d_ws, size_t ws_size,
                              hipStream_t stream) {
  (void)in_sizes; (void)n_in; (void)out_size; (void)ws_size;
  const float* x  = (const float*)d_in[0];
  // d_in[1] = causal tril mask (implied by the causal linear-attention path)
  const float* Wq = (const float*)d_in[2];
  const float* bq = (const float*)d_in[3];
  const float* Wk = (const float*)d_in[4];
  const float* bk = (const float*)d_in[5];
  const float* Wv = (const float*)d_in[6];
  const float* bv = (const float*)d_in[7];
  const float* Wo = (const float*)d_in[8];
  const float* bo = (const float*)d_in[9];
  float* out = (float*)d_out;

  char* ws = (char*)d_ws;
  const size_t tsz = (size_t)SEQ * DM * sizeof(float);  // 8 MB each
  float* Qh = (float*)(ws + 0 * tsz);  // head-major [H][N][64]
  float* Kh = (float*)(ws + 1 * tsz);
  float* Vh = (float*)(ws + 2 * tsz);
  float* Oh = (float*)(ws + 3 * tsz);

  dim3 g(SEQ / 128, DM / 64);
  gemm_bias_act<0, 1, 1><<<g, 256, 0, stream>>>(x,  Wq, bq, Qh, SEQ, DM, DM);
  gemm_bias_act<0, 1, 1><<<g, 256, 0, stream>>>(x,  Wk, bk, Kh, SEQ, DM, DM);
  gemm_bias_act<0, 0, 1><<<g, 256, 0, stream>>>(x,  Wv, bv, Vh, SEQ, DM, DM);
  linattn_causal<<<NH, 256, 0, stream>>>(Qh, Kh, Vh, Oh);
  gemm_bias_act<1, 0, 0><<<g, 256, 0, stream>>>(Oh, Wo, bo, out, SEQ, DM, DM);
}
